// CausalTemporalSelfAttention_18691697672766
// MI455X (gfx1250) — matrix operand order repaired and checked by reading, not compiled, not run on hardware
//
#include <hip/hip_runtime.h>

typedef __attribute__((ext_vector_type(16))) _Float16 v16h;
typedef __attribute__((ext_vector_type(8)))  _Float16 v8h;
typedef __attribute__((ext_vector_type(8)))  float    v8f;

constexpr int kB  = 4;
constexpr int kS  = 256;
constexpr int kN  = 32;
constexpr int kH  = 256;
constexpr int kNH = 8;
constexpr int kHD = 32;
constexpr int kFF = 512;
constexpr int kBN = kB * kN;       // 128
constexpr int kM  = kBN * kS;      // 32768 rows in [BN*S, H] activation matrices

#if defined(__has_builtin)
#  if __has_builtin(__builtin_amdgcn_tensor_load_to_lds)
#    define USE_TDM 1
#  endif
#endif
#ifndef USE_TDM
#  define USE_TDM 0
#endif

#if USE_TDM
typedef __attribute__((ext_vector_type(4))) unsigned int u32x4_t;
typedef __attribute__((ext_vector_type(4))) int          i32x4_t;
typedef __attribute__((ext_vector_type(8))) int          i32x8_t;

// TDM 2D tile load: 64 rows x 32 f16 (64B/row), row stride = strideElems*2 B.
// LDS pad: every 16 DWORDs insert 4 DWORDs -> row stride 40 halves (bank-safe),
// matching the [64][40] LDS tiles used by the WMMA fragment builders.
__device__ __forceinline__ void tdm_load_tile_64x32(
    const _Float16* gsrc, unsigned ldsOff, unsigned strideElems)
{
    const unsigned long long ga = (unsigned long long)(const void*)gsrc;
    u32x4_t g0; i32x8_t g1; i32x4_t g2; i32x4_t g3;
    g0[0] = 1u;                                   // count=1 (valid user D#)
    g0[1] = ldsOff;                               // lds_addr (bytes)
    g0[2] = (unsigned)ga;                         // global_addr[31:0]
    g0[3] = (unsigned)((ga >> 32) & 0x01ffffffull) | 0x80000000u; // [56:32]|type=2
    // data_size=2B, pad_enable, pad_interval=16dw, pad_amount=4dw
    g1[0] = (int)((1u << 16) | (1u << 20) | (3u << 22) | (3u << 25));
    g1[1] = (int)((strideElems & 0xffffu) << 16);            // tensor_dim0 lo16
    g1[2] = (int)((strideElems >> 16) & 0xffffu);            // dim0 hi | dim1 lo(0)
    g1[3] = (int)(0x4000u | (32u << 16));                    // dim1 hi (1<<30) | tile_dim0=32
    g1[4] = 64;                                              // tile_dim1=64, tile_dim2=0
    g1[5] = (int)strideElems;                                // tensor_dim0_stride lo32
    g1[6] = 0;                                               // stride hi | dim1_stride lo
    g1[7] = 0;
    g2[0] = 0; g2[1] = 0; g2[2] = 0; g2[3] = 0;
    g3[0] = 0; g3[1] = 0; g3[2] = 0; g3[3] = 0;
#if __clang_major__ >= 23
    i32x8_t g4;
    g4[0] = 0; g4[1] = 0; g4[2] = 0; g4[3] = 0;
    g4[4] = 0; g4[5] = 0; g4[6] = 0; g4[7] = 0;
    __builtin_amdgcn_tensor_load_to_lds(g0, g1, g2, g3, g4, 0);
#else
    __builtin_amdgcn_tensor_load_to_lds(g0, g1, g2, g3, 0);
#endif
}
#endif  // USE_TDM

// ---------------------------------------------------------------------------
// prep: build flat (f32 + f16), time_state (f16), flat+ts (f16), valid bytes
// ---------------------------------------------------------------------------
__global__ __launch_bounds__(256) void prep_kernel(
    const float* __restrict__ seq, const float* __restrict__ reg,
    const float* __restrict__ trs, const float* __restrict__ sess_emb,
    const float* __restrict__ reg_w, const float* __restrict__ reg_b,
    const float* __restrict__ tr_w, const float* __restrict__ tr_b,
    const unsigned char* __restrict__ vmask, const long long* __restrict__ codes,
    float* __restrict__ flat32, _Float16* __restrict__ X1,
    _Float16* __restrict__ Xv, _Float16* __restrict__ Xt,
    unsigned char* __restrict__ valid)
{
    const int r = blockIdx.x;          // r = bn*S + s
    const int c = threadIdx.x;
    const int bn = r >> 8, s = r & 255;
    const int b = bn >> 5, n = bn & 31;
    const int bs = b * kS + s;

    long long code = codes[bs];
    if (code < 0) code = 0;
    if (code > 7) code = 7;
    const float ts = sess_emb[(int)code * kH + c]
                   + reg[bs] * reg_w[c] + reg_b[c]
                   + trs[bs] * tr_w[c] + tr_b[c];
    const float sv = seq[((size_t)bs * kN + n) * kH + c];

    const size_t idx = (size_t)r * kH + c;
    flat32[idx] = sv;
    X1[idx] = (_Float16)(sv + ts);
    Xv[idx] = (_Float16)sv;
    Xt[idx] = (_Float16)ts;
    if (c == 0) valid[r] = vmask[(size_t)bs * kN + n];
}

__global__ __launch_bounds__(256) void allinv_kernel(
    const unsigned char* __restrict__ valid, unsigned char* __restrict__ ainv)
{
    __shared__ int red[256];
    const int c = threadIdx.x;
    red[c] = valid[blockIdx.x * kS + c];
    __syncthreads();
    for (int s = 128; s > 0; s >>= 1) {
        if (c < s) red[c] |= red[c + s];
        __syncthreads();
    }
    if (c == 0) ainv[blockIdx.x] = (red[0] == 0) ? 1 : 0;
}

// One-shot weight transpose+convert: W[K][N] f32 -> Wh[N][K] f16.
__global__ __launch_bounds__(256) void wcvt_kernel(
    const float* __restrict__ W, _Float16* __restrict__ Wh, int Ksz, int Nsz)
{
    const int idx = blockIdx.x * 256 + threadIdx.x;
    const int k = idx / Nsz, n = idx % Nsz;
    Wh[(size_t)n * Ksz + k] = (_Float16)W[idx];
}

// ---------------------------------------------------------------------------
// WMMA GEMM:  out = act((A_f16[M,K] @ Wh_f16[N,K]^T + bias) * scale)
// 128 threads (4 waves), 64x64 block tile, 32x32 per wave, K-step 32.
// Both operand tiles are 64 rows x 32 f16 staged to LDS; with TDM available
// the staging is an async double-buffered tensor_load_to_lds pipeline.
// ---------------------------------------------------------------------------
__global__ __launch_bounds__(128) void wmma_gemm(
    const _Float16* __restrict__ A, const _Float16* __restrict__ Wh,
    const float* __restrict__ bias,
    _Float16* __restrict__ outH, float* __restrict__ outF,
    int Nsz, int Ksz, float scaleOut, int doGelu)
{
    __shared__ __align__(16) _Float16 As[2][64][40];
    __shared__ __align__(16) _Float16 Ws[2][64][40];
    const int tid = threadIdx.x;
    const int wid = tid >> 5, lane = tid & 31;
    const int hf = lane >> 4, l15 = lane & 15;
    const int wm = (wid >> 1) * 32, wn = (wid & 1) * 32;
    const int m0 = blockIdx.y * 64, n0 = blockIdx.x * 64;
    const int nsteps = Ksz / 32;

    v8f acc[2][2];
    for (int i = 0; i < 2; ++i)
        for (int j = 0; j < 2; ++j)
            for (int r = 0; r < 8; ++r) acc[i][j][r] = 0.0f;

#if USE_TDM
    const unsigned asOff = (unsigned)(unsigned long long)(const void*)&As[0][0][0];
    const unsigned wsOff = (unsigned)(unsigned long long)(const void*)&Ws[0][0][0];
    const unsigned bufBytes = 64u * 40u * 2u;   // 5120
    if (wid == 0) {
        tdm_load_tile_64x32(A  + (size_t)m0 * Ksz, asOff, (unsigned)Ksz);
        tdm_load_tile_64x32(Wh + (size_t)n0 * Ksz, wsOff, (unsigned)Ksz);
    }
#endif

    for (int st = 0; st < nsteps; ++st) {
        const int cur = st & 1;
#if USE_TDM
        if (wid == 0) {
            if (st + 1 < nsteps) {
                const int k0n = (st + 1) * 32;
                const unsigned nb = (unsigned)(cur ^ 1) * bufBytes;
                tdm_load_tile_64x32(A  + (size_t)m0 * Ksz + k0n, asOff + nb, (unsigned)Ksz);
                tdm_load_tile_64x32(Wh + (size_t)n0 * Ksz + k0n, wsOff + nb, (unsigned)Ksz);
                __builtin_amdgcn_s_wait_tensorcnt(2);   // current tile retired
            } else {
                __builtin_amdgcn_s_wait_tensorcnt(0);
            }
        }
        __syncthreads();
#else
        {   // synchronous staging fallback: 16 f16 per thread per tile
            const int row = tid >> 1, cb = (tid & 1) * 16;
            const int k0 = st * 32;
            const _Float16* sa = A + (size_t)(m0 + row) * Ksz + k0 + cb;
            *(v8h*)&As[cur][row][cb]     = *(const v8h*)sa;
            *(v8h*)&As[cur][row][cb + 8] = *(const v8h*)(sa + 8);
            const _Float16* sw = Wh + (size_t)(n0 + row) * Ksz + k0 + cb;
            *(v8h*)&Ws[cur][row][cb]     = *(const v8h*)sw;
            *(v8h*)&Ws[cur][row][cb + 8] = *(const v8h*)(sw + 8);
        }
        __syncthreads();
#endif
        v16h af[2], bf[2];
        for (int ms = 0; ms < 2; ++ms) {
            // A lane layout: lane=row, elems 0-7: K=hf*8.., elems 8-15: K=16+hf*8..
            const _Float16* sp = &As[cur][wm + ms * 16 + l15][hf * 8];
            v8h lo = *(const v8h*)sp, hi = *(const v8h*)(sp + 16);
            for (int i = 0; i < 8; ++i) { af[ms][i] = lo[i]; af[ms][i + 8] = hi[i]; }
        }
        for (int ns = 0; ns < 2; ++ns) {
            // B lane layout: lane=col, elems 0-7: K=hf*8.., elems 8-15: K=16+hf*8..
            const _Float16* sp = &Ws[cur][wn + ns * 16 + l15][hf * 8];
            v8h lo = *(const v8h*)sp, hi = *(const v8h*)(sp + 16);
            for (int i = 0; i < 8; ++i) { bf[ns][i] = lo[i]; bf[ns][i + 8] = hi[i]; }
        }
        for (int ms = 0; ms < 2; ++ms)
            for (int ns = 0; ns < 2; ++ns)
                acc[ms][ns] = __builtin_amdgcn_wmma_f32_16x16x32_f16(
                    false, af[ms], false, bf[ns], (short)0, acc[ms][ns], false, false);
        __syncthreads();
    }

    for (int ms = 0; ms < 2; ++ms)
        for (int ns = 0; ns < 2; ++ns) {
            const int col = n0 + wn + ns * 16 + l15;
            const float bv = bias[col];
            for (int r = 0; r < 8; ++r) {
                const int row = m0 + wm + ms * 16 + hf * 8 + r;
                float v = (acc[ms][ns][r] + bv) * scaleOut;
                if (doGelu) v = 0.5f * v * (1.0f + erff(v * 0.70710678118654752f));
                const size_t oidx = (size_t)row * Nsz + col;
                if (outH) outH[oidx] = (_Float16)v;
                else      outF[oidx] = v;
            }
        }
}

// ---------------------------------------------------------------------------
// Flash attention per (bn, head). Online softmax tracks both the full softmax
// denominator (lrun) and the valid-masked sum (vrun) so the reference's
// post-softmax renormalization is exact: out = acc / max(vrun, 1e-6*lrun).
// ---------------------------------------------------------------------------
__global__ __launch_bounds__(128) void attn_flash(
    const _Float16* __restrict__ Q, const _Float16* __restrict__ Kd,
    const _Float16* __restrict__ V, const _Float16* __restrict__ TQ,
    const _Float16* __restrict__ TK,
    const unsigned char* __restrict__ valid,
    const unsigned char* __restrict__ allinv,
    _Float16* __restrict__ attnP)
{
    __shared__ __align__(16) _Float16 vt[kHD][kS];     // V transposed: vt[d][key]
    __shared__ __align__(16) _Float16 pbuf[4][16][32]; // per-wave prob tile
    __shared__ unsigned char vld[kS];

    const int tid = threadIdx.x;
    const int wid = tid >> 5, lane = tid & 31;
    const int hf = lane >> 4, l15 = lane & 15;
    const int bn = blockIdx.x >> 3, hh = blockIdx.x & 7;
    const size_t base = (size_t)bn * kS * kH + hh * kHD;

    for (int i = 0; i < 64; ++i) {                     // stage V^T (coalesced)
        const int idx = tid + 128 * i;
        const int key = idx >> 5, d = idx & 31;
        vt[d][key] = V[base + (size_t)key * kH + d];
    }
    vld[tid]       = valid[bn * kS + tid];
    vld[tid + 128] = valid[bn * kS + tid + 128];
    __syncthreads();

    const bool ainv = allinv[bn] != 0;
    const float rs = 0.17677669529663687f;             // 1/sqrt(32)

    for (int qi = 0; qi < 4; ++qi) {
        const int qt = wid + 4 * qi;
        v16h qf, tqf;
        {
            const _Float16* sp = Q + base + (size_t)(qt * 16 + l15) * kH + hf * 8;
            v8h lo = *(const v8h*)sp, hi = *(const v8h*)(sp + 16);
            for (int i = 0; i < 8; ++i) { qf[i] = lo[i]; qf[i + 8] = hi[i]; }
            sp = TQ + base + (size_t)(qt * 16 + l15) * kH + hf * 8;
            lo = *(const v8h*)sp; hi = *(const v8h*)(sp + 16);
            for (int i = 0; i < 8; ++i) { tqf[i] = lo[i]; tqf[i + 8] = hi[i]; }
        }
        float mrun[8], lrun[8], vrun[8];
        v8f oacc[2];
        for (int r = 0; r < 8; ++r) {
            mrun[r] = -1e30f; lrun[r] = 0.f; vrun[r] = 0.f;
            oacc[0][r] = 0.f; oacc[1][r] = 0.f;
        }

        for (int kp = 0; kp < 8; ++kp) {               // 32 keys per pass
            v8f sc0, sc1;
            for (int r = 0; r < 8; ++r) { sc0[r] = 0.f; sc1[r] = 0.f; }
            for (int half = 0; half < 2; ++half) {
                const int kt = kp * 2 + half;
                v16h kf, tkf;
                const _Float16* sp = Kd + base + (size_t)(kt * 16 + l15) * kH + hf * 8;
                v8h lo = *(const v8h*)sp, hi = *(const v8h*)(sp + 16);
                for (int i = 0; i < 8; ++i) { kf[i] = lo[i]; kf[i + 8] = hi[i]; }
                sp = TK + base + (size_t)(kt * 16 + l15) * kH + hf * 8;
                lo = *(const v8h*)sp; hi = *(const v8h*)(sp + 16);
                for (int i = 0; i < 8; ++i) { tkf[i] = lo[i]; tkf[i + 8] = hi[i]; }
                if (half == 0) {
                    sc0 = __builtin_amdgcn_wmma_f32_16x16x32_f16(false, tqf, false, tkf, (short)0, sc0, false, false);
                    sc0 = __builtin_amdgcn_wmma_f32_16x16x32_f16(false, qf,  false, kf,  (short)0, sc0, false, false);
                } else {
                    sc1 = __builtin_amdgcn_wmma_f32_16x16x32_f16(false, tqf, false, tkf, (short)0, sc1, false, false);
                    sc1 = __builtin_amdgcn_wmma_f32_16x16x32_f16(false, qf,  false, kf,  (short)0, sc1, false, false);
                }
            }
            const int key0 = kp * 32 + l15;
            const int key1 = key0 + 16;
            const float vm0 = (float)vld[key0];
            const float vm1 = (float)vld[key1];
            for (int r = 0; r < 8; ++r) {
                const int q = qt * 16 + hf * 8 + r;
                float sv0 = sc0[r] * rs;
                float sv1 = sc1[r] * rs;
                if (key0 > q || vld[key0] == 0) sv0 = -10000.0f;
                if (key1 > q || vld[key1] == 0) sv1 = -10000.0f;
                if (ainv && key0 == 0) sv0 = 0.0f;     // all-invalid -> first key 0

                float rm = fmaxf(sv0, sv1);
                for (int off = 1; off < 16; off <<= 1)
                    rm = fmaxf(rm, __shfl_xor(rm, off, 32));
                const float mnew = fmaxf(mrun[r], rm);
                const float fac = expf(mrun[r] - mnew);
                const float p0 = expf(sv0 - mnew);
                const float p1 = expf(sv1 - mnew);
                const float pv0 = p0 * vm0, pv1 = p1 * vm1;
                float ps = p0 + p1, pvs = pv0 + pv1;
                for (int off = 1; off < 16; off <<= 1) {
                    ps  += __shfl_xor(ps, off, 32);
                    pvs += __shfl_xor(pvs, off, 32);
                }
                lrun[r] = lrun[r] * fac + ps;
                vrun[r] = vrun[r] * fac + pvs;
                mrun[r] = mnew;
                oacc[0][r] *= fac;
                oacc[1][r] *= fac;
                pbuf[wid][hf * 8 + r][l15]      = (_Float16)pv0;
                pbuf[wid][hf * 8 + r][16 + l15] = (_Float16)pv1;
            }
            __syncthreads();
            v16h af;
            {
                const _Float16* sp = &pbuf[wid][l15][hf * 8];
                v8h lo = *(const v8h*)sp, hi = *(const v8h*)(sp + 16);
                for (int i = 0; i < 8; ++i) { af[i] = lo[i]; af[i + 8] = hi[i]; }
            }
            const int kb = kp * 32;
            for (int nn = 0; nn < 2; ++nn) {
                v16h bfv;
                const _Float16* sp = &vt[nn * 16 + l15][kb + hf * 8];
                v8h lo = *(const v8h*)sp, hi = *(const v8h*)(sp + 16);
                for (int i = 0; i < 8; ++i) { bfv[i] = lo[i]; bfv[i + 8] = hi[i]; }
                oacc[nn] = __builtin_amdgcn_wmma_f32_16x16x32_f16(
                    false, af, false, bfv, (short)0, oacc[nn], false, false);
            }
            __syncthreads();
        }

        for (int nn = 0; nn < 2; ++nn) {
            const int d = hh * kHD + nn * 16 + l15;
            for (int r = 0; r < 8; ++r) {
                const int q = qt * 16 + hf * 8 + r;
                const float den = fmaxf(vrun[r], 1e-6f * lrun[r]);
                const float ov = ainv ? 0.0f : oacc[nn][r] / den;
                attnP[(size_t)(bn * kS + q) * kH + d] = (_Float16)ov;
            }
        }
    }
}

// ---------------------------------------------------------------------------
__global__ __launch_bounds__(256) void ln_add_kernel(
    const float* __restrict__ Xa, const float* __restrict__ Xb,
    const float* __restrict__ g, const float* __restrict__ bet,
    float* __restrict__ o32, _Float16* __restrict__ o16)
{
    __shared__ float red[256];
    const int r = blockIdx.x, c = threadIdx.x;
    const size_t idx = (size_t)r * kH + c;
    const float v = Xa[idx] + Xb[idx];
    red[c] = v; __syncthreads();
    for (int s = 128; s > 0; s >>= 1) { if (c < s) red[c] += red[c + s]; __syncthreads(); }
    const float mean = red[0] * (1.0f / kH);
    __syncthreads();
    const float dd = v - mean;
    red[c] = dd * dd; __syncthreads();
    for (int s = 128; s > 0; s >>= 1) { if (c < s) red[c] += red[c + s]; __syncthreads(); }
    const float var = red[0] * (1.0f / kH);
    const float xn = dd * rsqrtf(var + 1e-5f) * g[c] + bet[c];
    o32[idx] = xn;
    o16[idx] = (_Float16)xn;
}

__global__ __launch_bounds__(256) void ln2_final_kernel(
    const float* __restrict__ X, const float* __restrict__ FF,
    const float* __restrict__ g, const float* __restrict__ bet,
    const unsigned char* __restrict__ valid, float* __restrict__ out)
{
    __shared__ float red[256];
    const int r = blockIdx.x, c = threadIdx.x;
    const int bn = r >> 8, s = r & 255;
    const int b = bn >> 5, n = bn & 31;
    const size_t idx = (size_t)r * kH + c;
    const float v = X[idx] + FF[idx];
    red[c] = v; __syncthreads();
    for (int t = 128; t > 0; t >>= 1) { if (c < t) red[c] += red[c + t]; __syncthreads(); }
    const float mean = red[0] * (1.0f / kH);
    __syncthreads();
    const float dd = v - mean;
    red[c] = dd * dd; __syncthreads();
    for (int t = 128; t > 0; t >>= 1) { if (c < t) red[c] += red[c + t]; __syncthreads(); }
    const float var = red[0] * (1.0f / kH);
    float xn = dd * rsqrtf(var + 1e-5f) * g[c] + bet[c];
    xn *= (float)valid[r];
    out[(((size_t)b * kS + s) * kN + n) * kH + c] = xn;
}

// ---------------------------------------------------------------------------
extern "C" void kernel_launch(void* const* d_in, const int* in_sizes, int n_in,
                              void* d_out, int out_size, void* d_ws, size_t ws_size,
                              hipStream_t stream)
{
    (void)in_sizes; (void)n_in; (void)out_size; (void)ws_size;
    const float* seq   = (const float*)d_in[0];
    const float* reg   = (const float*)d_in[1];
    const float* trs   = (const float*)d_in[2];
    const float* q_w = (const float*)d_in[3];  const float* q_b = (const float*)d_in[4];
    const float* k_w = (const float*)d_in[5];  const float* k_b = (const float*)d_in[6];
    const float* v_w = (const float*)d_in[7];  const float* v_b = (const float*)d_in[8];
    const float* o_w = (const float*)d_in[9];  const float* o_b = (const float*)d_in[10];
    const float* sess_emb = (const float*)d_in[11];
    const float* reg_w = (const float*)d_in[12]; const float* reg_b = (const float*)d_in[13];
    const float* tr_w  = (const float*)d_in[14]; const float* tr_b  = (const float*)d_in[15];
    const float* tq_w = (const float*)d_in[16]; const float* tq_b = (const float*)d_in[17];
    const float* tk_w = (const float*)d_in[18]; const float* tk_b = (const float*)d_in[19];
    const float* ff_w1 = (const float*)d_in[20]; const float* ff_b1 = (const float*)d_in[21];
    const float* ff_w2 = (const float*)d_in[22]; const float* ff_b2 = (const float*)d_in[23];
    const float* ln1_g = (const float*)d_in[24]; const float* ln1_b = (const float*)d_in[25];
    const float* ln2_g = (const float*)d_in[26]; const float* ln2_b = (const float*)d_in[27];
    const unsigned char* vmask = (const unsigned char*)d_in[28];
    const long long* codes = (const long long*)d_in[29];
    float* out = (float*)d_out;
    char* ws = (char*)d_ws;

    const size_t F32 = (size_t)kM * kH * 4;   // 33,554,432 B
    const size_t F16 = (size_t)kM * kH * 2;   // 16,777,216 B
    float*    flat32 = (float*)   (ws + 0);
    _Float16* X1     = (_Float16*)(ws + F32);
    _Float16* Xv     = (_Float16*)(ws + F32 + F16);
    _Float16* Xt     = (_Float16*)(ws + F32 + 2 * F16);
    _Float16* Qh     = (_Float16*)(ws + F32 + 3 * F16);
    _Float16* Kh     = (_Float16*)(ws + F32 + 4 * F16);
    _Float16* Vh     = (_Float16*)(ws + F32 + 5 * F16);
    _Float16* TQh    = (_Float16*)(ws + F32 + 6 * F16);
    _Float16* TKh    = (_Float16*)(ws + F32 + 7 * F16);
    _Float16* G16    = (_Float16*)(ws + F32 + 8 * F16);          // [M, FF] f16
    unsigned char* validp = (unsigned char*)(ws + 2 * F32 + 8 * F16);
    unsigned char* ainvp  = validp + kM;
    // f16 pre-transposed weights ([N][K] layout), 1.31 MB total
    _Float16* whBase = (_Float16*)(ws + 2 * F32 + 8 * F16 + 65536);
    _Float16* qwh  = whBase;
    _Float16* kwh  = whBase + 65536;
    _Float16* vwh  = whBase + 131072;
    _Float16* owh  = whBase + 196608;
    _Float16* tqwh = whBase + 262144;
    _Float16* tkwh = whBase + 327680;
    _Float16* f1wh = whBase + 393216;   // [512][256]
    _Float16* f2wh = whBase + 524288;   // [256][512]
    // aliases (producer runs strictly after last consumer of the aliased buffer)
    _Float16* attnP = X1;                 // attention output (f16) over X1
    float*    attnO = (float*)Xv;         // O-proj f32 over Xv+Xt
    float*    x32   = (float*)Qh;         // LN1 f32 over Q+K
    _Float16* x16   = TQh;                // LN1 f16 over TQ
    float*    ff2o  = flat32;             // FF2 f32 over flat32

    prep_kernel<<<kM, 256, 0, stream>>>(seq, reg, trs, sess_emb, reg_w, reg_b,
                                        tr_w, tr_b, vmask, codes,
                                        flat32, X1, Xv, Xt, validp);
    allinv_kernel<<<kBN, 256, 0, stream>>>(validp, ainvp);

    wcvt_kernel<<<256, 256, 0, stream>>>(q_w,  qwh,  kH,  kH);
    wcvt_kernel<<<256, 256, 0, stream>>>(k_w,  kwh,  kH,  kH);
    wcvt_kernel<<<256, 256, 0, stream>>>(v_w,  vwh,  kH,  kH);
    wcvt_kernel<<<256, 256, 0, stream>>>(o_w,  owh,  kH,  kH);
    wcvt_kernel<<<256, 256, 0, stream>>>(tq_w, tqwh, kH,  kH);
    wcvt_kernel<<<256, 256, 0, stream>>>(tk_w, tkwh, kH,  kH);
    wcvt_kernel<<<512, 256, 0, stream>>>(ff_w1, f1wh, kH,  kFF);  // -> [512][256]
    wcvt_kernel<<<512, 256, 0, stream>>>(ff_w2, f2wh, kFF, kH);   // -> [256][512]

    const dim3 gBlk(128);
    wmma_gemm<<<dim3(kH / 64, kM / 64), gBlk, 0, stream>>>(X1, qwh,  q_b,  Qh,  nullptr, kH,  kH,  1.0f,  0);
    wmma_gemm<<<dim3(kH / 64, kM / 64), gBlk, 0, stream>>>(X1, kwh,  k_b,  Kh,  nullptr, kH,  kH,  1.0f,  0);
    wmma_gemm<<<dim3(kH / 64, kM / 64), gBlk, 0, stream>>>(Xv, vwh,  v_b,  Vh,  nullptr, kH,  kH,  1.0f,  0);
    wmma_gemm<<<dim3(kH / 64, kM / 64), gBlk, 0, stream>>>(Xt, tqwh, tq_b, TQh, nullptr, kH,  kH,  0.25f, 0);
    wmma_gemm<<<dim3(kH / 64, kM / 64), gBlk, 0, stream>>>(Xt, tkwh, tk_b, TKh, nullptr, kH,  kH,  1.0f,  0);

    attn_flash<<<kBN * kNH, gBlk, 0, stream>>>(Qh, Kh, Vh, TQh, TKh, validp, ainvp, attnP);

    wmma_gemm<<<dim3(kH / 64, kM / 64), gBlk, 0, stream>>>(attnP, owh, o_b, nullptr, attnO, kH, kH, 1.0f, 0);
    ln_add_kernel<<<kM, 256, 0, stream>>>(flat32, attnO, ln1_g, ln1_b, x32, x16);

    wmma_gemm<<<dim3(kFF / 64, kM / 64), gBlk, 0, stream>>>(x16, f1wh, ff_b1, G16, nullptr, kFF, kH,  1.0f, 1);
    wmma_gemm<<<dim3(kH  / 64, kM / 64), gBlk, 0, stream>>>(G16, f2wh, ff_b2, nullptr, ff2o, kH,  kFF, 1.0f, 0);

    ln2_final_kernel<<<kM, 256, 0, stream>>>(x32, ff2o, ln2_g, ln2_b, validp, out);
}
